// SparseGuidedConv2d_86646670229785
// MI455X (gfx1250) — matrix operand order repaired and checked by reading, not compiled, not run on hardware
//
#include <hip/hip_runtime.h>
#include <hip/hip_bf16.h>

typedef __attribute__((ext_vector_type(16))) _Float16 v16h;
typedef __attribute__((ext_vector_type(8)))  float    v8f;

#define BN   4
#define CN   32
#define CGN  8
#define HN   256
#define WN   256
#define HWC  (HN*WN)
#define EPSC 1e-5f

#define TILES_PER_ROW (WN/16)            // 16
#define NT  (BN*HN*TILES_PER_ROW)        // 16384 tiles of 16 pixels
#define NWAVE 4
#define GRID  1024
#define BLOCK 128

// xs/gs halo tiles: [chan][3 rows][18 cols], row-major, col 0 == w0-1
#define XRC 54   // 3*18 floats per channel

__global__ __launch_bounds__(BLOCK) void pac_wmma_kernel(
    const float* __restrict__ x, const unsigned char* __restrict__ mask,
    const float* __restrict__ guide, const float* __restrict__ weight,
    const float* __restrict__ bias, float* __restrict__ out)
{
  // B fragments: [tap][ntile][lane][16 halves]; lane<16 -> K(c)=0..15, lane>=16 -> K=16..31
  __shared__ __align__(32) _Float16 wB[9*2*32*16];    // 18432 B
  __shared__ float wsumS[32*9];                       //  1152 B
  __shared__ float biasS[32];                         //   128 B
  __shared__ float xsA[NWAVE][32*XRC];                // 27648 B (masked x halo)
  __shared__ float gsA[NWAVE][CGN*XRC];               //  6912 B (guide halo)
  __shared__ float msA[NWAVE][XRC];                   //   864 B (mask halo)
  __shared__ float ksA[NWAVE][9*16];                  //  2304 B (kern[tap][pixel])

  const int tid  = threadIdx.x;
  const int wave = tid >> 5;
  const int lane = tid & 31;
  const int hi   = lane >> 4;       // half-wave select
  const int lp   = lane & 15;       // pixel row M for A, channel column N for B/C/D

  // ---- stage weights into WMMA B-fragment layout + wsum + bias ----
  for (int idx = tid; idx < 9*2*32*16; idx += BLOCK) {
    int j   = idx & 15;
    int ln  = (idx >> 4) & 31;
    int nt  = (idx >> 9) & 1;
    int tap = idx >> 10;
    int o = nt*16 + (ln & 15);
    int c = ((ln >> 4) << 4) + j;                 // K index = c
    wB[idx] = (_Float16)weight[(o*CN + c)*9 + tap];
  }
  for (int idx = tid; idx < 32*9; idx += BLOCK) {
    int o = idx / 9, t = idx % 9;
    float s = 0.f;
    for (int c = 0; c < CN; ++c) s += weight[(o*CN + c)*9 + t];
    wsumS[idx] = s;
  }
  if (tid < 32) biasS[tid] = bias[tid];
  __syncthreads();

  float* xs = xsA[wave];
  float* gs = gsA[wave];
  float* ms = msA[wave];
  float* ks = ksA[wave];

  for (int tile = blockIdx.x*NWAVE + wave; tile < NT; tile += GRID*NWAVE) {
    const int b  = tile >> 12;           // / 4096 tiles per batch
    const int rm = tile & 4095;
    const int h  = rm >> 4;
    const int w0 = (rm & 15) << 4;

    // ---- stage mask halo (zero padded, as reference's zero-pad unfold) ----
    const unsigned char* mbase = mask + b*HWC;
    for (int idx = lane; idx < XRC; idx += 32) {
      int r = idx / 18, col = idx % 18;
      int hh = h + r - 1, ww = w0 + col - 1;
      float m = 0.f;
      if (hh >= 0 && hh < HN && ww >= 0 && ww < WN)
        m = mbase[hh*WN + ww] ? 1.f : 0.f;
      ms[idx] = m;
    }
    // ---- stage guide halo ----
    for (int idx = lane; idx < CGN*XRC; idx += 32) {
      int cg = idx / XRC, rc = idx % XRC;
      int r = rc / 18, col = rc % 18;
      int hh = h + r - 1, ww = w0 + col - 1;
      float g = 0.f;
      if (hh >= 0 && hh < HN && ww >= 0 && ww < WN)
        g = guide[(b*CGN + cg)*HWC + hh*WN + ww];
      gs[idx] = g;
    }
    // ---- stage masked x halo (xm = x * mask; OOB -> 0) ----
    for (int idx = lane; idx < CN*XRC; idx += 32) {
      int c = idx / XRC, rc = idx % XRC;
      int r = rc / 18, col = rc % 18;
      int hh = h + r - 1, ww = w0 + col - 1;
      float v = 0.f;
      if (hh >= 0 && hh < HN && ww >= 0 && ww < WN)
        v = x[(b*CN + c)*HWC + hh*WN + ww] * ms[rc];
      xs[idx] = v;
    }

    // ---- per-pixel PAC kernel weights: kern = exp(-0.5*||g_tap - g_ctr||^2) ----
    for (int t = 5*hi; t < (hi ? 9 : 5); ++t) {     // hi=0: taps 0..4, hi=1: taps 5..8
      int di = t / 3, dj = t % 3;
      float s = 0.f;
      #pragma unroll
      for (int cg = 0; cg < CGN; ++cg) {
        float d = gs[cg*XRC + di*18 + lp + dj] - gs[cg*XRC + 18 + lp + 1];
        s += d * d;
      }
      ks[t*16 + lp] = __expf(-0.5f * s);
    }
    float kv[9];
    #pragma unroll
    for (int t = 0; t < 9; ++t) kv[t] = ks[t*16 + lp];

    // ---- implicit GEMM: 9 taps x (16 px  x 32 out-ch), f16 WMMA, f32 acc ----
    v8f acc0 = {}; v8f acc1 = {};
    #pragma unroll
    for (int t = 0; t < 9; ++t) {
      const int di = t / 3, dj = t % 3;
      const float sc = kv[t];
      v16h a;
      #pragma unroll
      for (int v = 0; v < 8; ++v) {
        // contiguous k order, same as wB: lane-half hi covers K = 0..15 (+16 for hi=1)
        int c0 = 2*v + 16*hi;
        a[2*v]   = (_Float16)(xs[ c0      *XRC + di*18 + lp + dj] * sc);
        a[2*v+1] = (_Float16)(xs[(c0 + 1) *XRC + di*18 + lp + dj] * sc);
      }
      const v16h b0 = *(const v16h*)&wB[(t*2 + 0)*512 + lane*16];
      const v16h b1 = *(const v16h*)&wB[(t*2 + 1)*512 + lane*16];
      acc0 = __builtin_amdgcn_wmma_f32_16x16x32_f16(false, a, false, b0,
                                                    (short)0, acc0, false, false);
      acc1 = __builtin_amdgcn_wmma_f32_16x16x32_f16(false, a, false, b1,
                                                    (short)0, acc1, false, false);
    }

    // ---- mask-normalizer (VALU), bias, divide, mask passthrough ----
    const int o0 = lp, o1 = 16 + lp;     // C/D layout: N = lane&15 (+ntile*16)
    float ws0[9], ws1[9];
    #pragma unroll
    for (int t = 0; t < 9; ++t) { ws0[t] = wsumS[o0*9 + t]; ws1[t] = wsumS[o1*9 + t]; }
    const float bb0 = biasS[o0], bb1 = biasS[o1];

    float outv0[8], outv1[8];
    #pragma unroll
    for (int r = 0; r < 8; ++r) {
      int M = 8*hi + r;                  // C/D layout: M = r + 8*(lane/16)
      float mc0 = 0.f, mc1 = 0.f;
      #pragma unroll
      for (int t = 0; t < 9; ++t) {
        int di = t / 3, dj = t % 3;
        float mk = ms[di*18 + M + dj] * ks[t*16 + M];
        mc0 += mk * ws0[t];
        mc1 += mk * ws1[t];
      }
      float r0 = (acc0[r] + bb0) / (mc0 + bb0 + EPSC);
      float r1 = (acc1[r] + bb1) / (mc1 + bb1 + EPSC);
      if (ms[18 + M + 1] > 0.5f) {       // mask true -> passthrough x (xs==x there)
        r0 = xs[o0*XRC + 18 + M + 1];
        r1 = xs[o1*XRC + 18 + M + 1];
      }
      outv0[r] = r0; outv1[r] = r1;
    }

    // ---- transpose through LDS (overlay xs; same-wave LDS ops are in-order) ----
    __builtin_amdgcn_wave_barrier();
    float* os = xs;                      // 512 floats: [o][M]
    #pragma unroll
    for (int r = 0; r < 8; ++r) {
      int M = 8*hi + r;
      os[o0*16 + M] = outv0[r];
      os[o1*16 + M] = outv1[r];
    }
    __builtin_amdgcn_wave_barrier();
    float* obase = out + (size_t)b*CN*HWC + h*WN + w0;
    for (int idx = lane; idx < 512; idx += 32) {
      int o = idx >> 4, M = idx & 15;    // 32 lanes = 2 channels x 64B contiguous
      obase[(size_t)o*HWC + M] = os[idx];
    }
  }
}

extern "C" void kernel_launch(void* const* d_in, const int* in_sizes, int n_in,
                              void* d_out, int out_size, void* d_ws, size_t ws_size,
                              hipStream_t stream) {
  const float*         x      = (const float*)d_in[0];
  const unsigned char* mask   = (const unsigned char*)d_in[1];  // bool array, 1B/elt
  const float*         guide  = (const float*)d_in[2];
  const float*         weight = (const float*)d_in[3];
  const float*         bias   = (const float*)d_in[4];
  float*               out    = (float*)d_out;
  (void)in_sizes; (void)n_in; (void)d_ws; (void)ws_size; (void)out_size;
  pac_wmma_kernel<<<GRID, BLOCK, 0, stream>>>(x, mask, guide, weight, bias, out);
}
